// CrossNetMixFuxiCTR_29583734734870
// MI455X (gfx1250) — matrix operand order repaired and checked by reading, not compiled, not run on hardware
//
#include <hip/hip_runtime.h>
#include <hip/hip_bf16.h>
#include <math.h>

// ---------------------------------------------------------------------------
// CrossNetMix collapses algebraically: Xi = s_b * X0 (per-row scalar s).
// => one GEMM  Y = X0 @ [U;V;Wg]^T, per-row scalar recurrence, out = s*X0.
// GEMM runs in split-bf16 (hi/lo, 3 WMMA products) on v_wmma_f32_16x16x32_bf16,
// with double-buffered GLOBAL_LOAD_ASYNC_TO_LDS_B128 staging (ASYNCcnt).
// ---------------------------------------------------------------------------

typedef __bf16        bf16_t;
typedef bf16_t        v16bf __attribute__((ext_vector_type(16)));
typedef float         v8f   __attribute__((ext_vector_type(8)));
typedef unsigned int  u32x4 __attribute__((ext_vector_type(4)));
typedef unsigned int  u32x8 __attribute__((ext_vector_type(8)));
typedef unsigned short ushort_t;

#define BB   16384      // batch
#define DD   2048       // feature dim
#define NP   3200       // padded N: 24 groups*128 (U|V) + 1 group (Wg pad)
#define NGRP 25
#define LDT  72         // padded K-stride (bf16 elems) for LDS tiles
#define LDC  132        // padded col-stride (f32) for C tile in LDS
#define BUFB 55296      // bytes per staging buffer

// ---- bf16 split helpers ----------------------------------------------------
static __device__ __forceinline__ ushort_t f2bf_rne(float x) {
    unsigned u = __builtin_bit_cast(unsigned, x);
    unsigned r = u + 0x7FFFu + ((u >> 16) & 1u);
    return (ushort_t)(r >> 16);
}
static __device__ __forceinline__ float bf2f(ushort_t h) {
    return __builtin_bit_cast(float, (unsigned)h << 16);
}

// ---- async 16B copy global -> LDS (ASYNCcnt tracked) ------------------------
static __device__ __forceinline__ void async_ld16(unsigned lds_addr, const void* gptr) {
    asm volatile("global_load_async_to_lds_b128 %0, %1, off"
                 :: "v"(lds_addr), "v"(gptr)
                 : "memory");
}

// ---- fragment loader: two 16B LDS chunks -> v16bf --------------------------
static __device__ __forceinline__ v16bf ld_frag(const ushort_t* p, int gap_elems) {
    u32x4 a = *(const u32x4*)p;
    u32x4 b = *(const u32x4*)(p + gap_elems);
    u32x8 c;
    c[0]=a[0]; c[1]=a[1]; c[2]=a[2]; c[3]=a[3];
    c[4]=b[0]; c[5]=b[1]; c[6]=b[2]; c[7]=b[3];
    return __builtin_bit_cast(v16bf, c);
}

static __device__ __forceinline__ v8f wmma_bf16(v16bf a, v16bf b, v8f c) {
    // (neg_a, A, neg_b, B, c_mod, C, reuse_a, reuse_b)
    return __builtin_amdgcn_wmma_f32_16x16x32_bf16(false, a, false, b,
                                                   (short)0, c, false, false);
}

// ---------------------------------------------------------------------------
// Kernel 1: split X0 (f32) into hi/lo bf16 planes.
// ---------------------------------------------------------------------------
__global__ __launch_bounds__(256) void k_split_x(const float* __restrict__ X0,
                                                 ushort_t* __restrict__ hi,
                                                 ushort_t* __restrict__ lo,
                                                 int n) {
    int i = blockIdx.x * 256 + threadIdx.x;
    if (i >= n) return;
    float x = X0[i];
    ushort_t h = f2bf_rne(x);
    hi[i] = h;
    lo[i] = f2bf_rne(x - bf2f(h));
}

// ---------------------------------------------------------------------------
// Kernel 2: build padded combined weight matrix W_cat (NP x D), hi/lo bf16.
// Row layout: n = g*128 + r2 for g<24 (g = l*8+e): r2<64 -> U rank, else V rank.
//             n = 3072 + (l*8+e) -> Wg row.  Rows 3096..3199 = 0.
// ---------------------------------------------------------------------------
__global__ __launch_bounds__(256) void k_build_w(const float* __restrict__ U,
                                                 const float* __restrict__ V,
                                                 const float* __restrict__ Wg,
                                                 ushort_t* __restrict__ Whi,
                                                 ushort_t* __restrict__ Wlo) {
    int n = blockIdx.x;
    const float* src = nullptr;
    if (n < 3072) {
        int grp = n >> 7, r2 = n & 127;          // grp = l*8+e
        size_t base = (size_t)grp * 64;
        src = (r2 < 64) ? U + (base + r2) * DD
                        : V + (base + (r2 - 64)) * DD;
    } else if (n < 3096) {
        src = Wg + (size_t)(n - 3072) * DD;
    }
    for (int d = threadIdx.x; d < DD; d += 256) {
        float x = src ? src[d] : 0.0f;
        ushort_t h = f2bf_rne(x);
        Whi[(size_t)n * DD + d] = h;
        Wlo[(size_t)n * DD + d] = f2bf_rne(x - bf2f(h));
    }
}

// ---------------------------------------------------------------------------
// Kernel 3: fused GEMM + rank-contraction.
// Block = 256 thr (8 waves, 4M x 2N). Tile: 64(M) x 128(N), K-step 64.
// Double-buffered async-to-LDS staging; 12 async b128 ops per wave per tile.
// Epilogue: group<24 -> P[b,g] = sum_r C[b,r]*C[b,64+r];  group 24 -> G[b,0..23].
// ---------------------------------------------------------------------------
__global__ __launch_bounds__(256) void k_gemm(const ushort_t* __restrict__ Xhi,
                                              const ushort_t* __restrict__ Xlo,
                                              const ushort_t* __restrict__ Whi,
                                              const ushort_t* __restrict__ Wlo,
                                              float* __restrict__ P,
                                              float* __restrict__ G) {
    __shared__ __align__(16) char smem[2 * BUFB];

    const int t    = threadIdx.x;
    const int lane = t & 31;
    const int wave = t >> 5;
    const int wm   = wave & 3;   // M quadrant (16 rows each)
    const int wn   = wave >> 2;  // N half (64 cols each)
    const int m0   = blockIdx.x * 64;
    const int g    = blockIdx.y;
    const int n0   = g * 128;

    // LDS byte address of smem: generic shared pointer carries LDS offset in [31:0]
    const unsigned lds_base = (unsigned)(unsigned long long)(uintptr_t)smem;

    // issue one tile's async loads (12 per wave: 2+2 X hi/lo, 4+4 W hi/lo)
    auto issue_tile = [&](int buf, int kb) {
        const unsigned bo = lds_base + buf * BUFB;
        #pragma unroll
        for (int i = 0; i < 2; ++i) {                 // X: 512 vecs / block
            int vid = t + i * 256;
            int row = vid >> 3, c8 = (vid & 7) * 8;
            size_t go = (size_t)(m0 + row) * DD + kb + c8;
            unsigned lo = (unsigned)(row * LDT + c8) * 2u;
            async_ld16(bo +        lo, Xhi + go);
            async_ld16(bo + 9216 + lo, Xlo + go);
        }
        #pragma unroll
        for (int i = 0; i < 4; ++i) {                 // W: 1024 vecs / block
            int vid = t + i * 256;
            int row = vid >> 3, c8 = (vid & 7) * 8;
            size_t go = (size_t)(n0 + row) * DD + kb + c8;
            unsigned lo = (unsigned)(row * LDT + c8) * 2u;
            async_ld16(bo + 18432 + lo, Whi + go);
            async_ld16(bo + 36864 + lo, Wlo + go);
        }
    };

    v8f acc[4] = {};

    issue_tile(0, 0);                                  // prologue
    int ibuf = 0;
    for (int kb = 0; kb < DD; kb += 64) {
        if (kb + 64 < DD) {
            issue_tile(ibuf ^ 1, kb + 64);             // prefetch next tile
            asm volatile("s_wait_asynccnt 0xc" ::: "memory");   // current 12 done
        } else {
            asm volatile("s_wait_asynccnt 0x0" ::: "memory");
        }
        __syncthreads();                               // all waves' tiles visible

        const char* buf = smem + ibuf * BUFB;
        const ushort_t* sXhi = (const ushort_t*)(buf);
        const ushort_t* sXlo = (const ushort_t*)(buf + 9216);
        const ushort_t* sWhi = (const ushort_t*)(buf + 18432);
        const ushort_t* sWlo = (const ushort_t*)(buf + 36864);

        #pragma unroll
        for (int ks = 0; ks < 64; ks += 32) {
            // A fragment: lane<16 -> M=lane, K=ks+{0..7,16..23}; lane>=16 -> K+8
            int am    = wm * 16 + (lane & 15);
            int akoff = ks + ((lane >> 4) << 3);
            v16bf ahi = ld_frag(sXhi + am * LDT + akoff, 16);
            v16bf alo = ld_frag(sXlo + am * LDT + akoff, 16);
            #pragma unroll
            for (int nf = 0; nf < 4; ++nf) {
                int bn = wn * 64 + nf * 16 + (lane & 15);
                int bk = ks + ((lane >> 4) << 3);     // K=ks+{0..7,16..23}; lane>=16 -> K+8
                v16bf bhi = ld_frag(sWhi + bn * LDT + bk, 16);
                v16bf blo = ld_frag(sWlo + bn * LDT + bk, 16);
                acc[nf] = wmma_bf16(ahi, bhi, acc[nf]);
                acc[nf] = wmma_bf16(ahi, blo, acc[nf]);
                acc[nf] = wmma_bf16(alo, bhi, acc[nf]);
            }
        }
        __syncthreads();                               // tile consumed; buffer reusable
        ibuf ^= 1;
    }

    // ---- dump C tile to LDS (f32 [64][LDC]) into buffer 0 ----
    float* Cs = (float*)smem;
    {
        int mofs = wm * 16 + ((lane >> 4) << 3);      // +8 for lanes 16..31
        int col0 = wn * 64 + (lane & 15);
        #pragma unroll
        for (int nf = 0; nf < 4; ++nf) {
            #pragma unroll
            for (int v = 0; v < 8; ++v)
                Cs[(mofs + v) * LDC + col0 + nf * 16] = acc[nf][v];
        }
    }
    __syncthreads();

    if (g < 24) {
        // P[b,g] = sum_{r<64} C[b,r] * C[b,64+r]  (4 threads per row)
        int row = t >> 2, q = t & 3;
        const float* cr = Cs + row * LDC;
        float p = 0.0f;
        #pragma unroll
        for (int r = 0; r < 16; ++r) {
            int rr = q * 16 + r;
            p += cr[rr] * cr[64 + rr];
        }
        p += __shfl_xor(p, 1);
        p += __shfl_xor(p, 2);
        if (q == 0) P[(size_t)(m0 + row) * 24 + g] = p;
    } else {
        for (int idx = t; idx < 64 * 24; idx += 256) {
            int row = idx / 24, col = idx % 24;
            G[(size_t)(m0 + row) * 24 + col] = Cs[row * LDC + col];
        }
    }
}

// ---------------------------------------------------------------------------
// Kernel 4: per-row scalar recurrence + out = s * X0.  One wave per row.
// ---------------------------------------------------------------------------
__global__ __launch_bounds__(256) void k_final(const float* __restrict__ X0,
                                               const float* __restrict__ P,
                                               const float* __restrict__ G,
                                               const float* __restrict__ bg,
                                               float* __restrict__ out) {
    int wave = threadIdx.x >> 5, lane = threadIdx.x & 31;
    int b = blockIdx.x * 8 + wave;
    const float* Pb = P + (size_t)b * 24;
    const float* Gb = G + (size_t)b * 24;

    float s = 1.0f;   // all lanes compute redundantly (broadcast loads)
    #pragma unroll
    for (int l = 0; l < 3; ++l) {
        float z[8], mx = -1e30f;
        #pragma unroll
        for (int e = 0; e < 8; ++e) {
            z[e] = s * Gb[l * 8 + e] + bg[l * 8 + e];
            mx = fmaxf(mx, z[e]);
        }
        float den = 0.0f, num = 0.0f, s2 = s * s;
        #pragma unroll
        for (int e = 0; e < 8; ++e) {
            float pe = __expf(z[e] - mx);
            den += pe;
            num += pe * (s2 * Pb[l * 8 + e]);
        }
        s += num / den;
    }

    const float4* x4 = (const float4*)(X0 + (size_t)b * DD);
    float4*       o4 = (float4*)(out + (size_t)b * DD);
    for (int i = lane; i < DD / 4; i += 32) {
        float4 v = x4[i];
        float4 r; r.x = v.x * s; r.y = v.y * s; r.z = v.z * s; r.w = v.w * s;
        o4[i] = r;
    }
}

// ---------------------------------------------------------------------------
// Workspace layout (bytes):
//   Xhi: 0            (B*D*2 = 67,108,864)
//   Xlo: 67,108,864
//   Whi: 134,217,728  (NP*D*2 = 13,107,200)
//   Wlo: 147,324,928
//   P  : 160,432,128  (B*24*4 = 1,572,864)
//   G  : 162,004,992
//   total: 163,577,856
// ---------------------------------------------------------------------------
extern "C" void kernel_launch(void* const* d_in, const int* in_sizes, int n_in,
                              void* d_out, int out_size, void* d_ws, size_t ws_size,
                              hipStream_t stream) {
    const float* X0 = (const float*)d_in[0];
    const float* U  = (const float*)d_in[1];
    const float* V  = (const float*)d_in[2];
    const float* Wg = (const float*)d_in[3];
    const float* bg = (const float*)d_in[4];
    float* out = (float*)d_out;

    char* ws = (char*)d_ws;
    ushort_t* Xhi = (ushort_t*)(ws);
    ushort_t* Xlo = (ushort_t*)(ws + 67108864);
    ushort_t* Whi = (ushort_t*)(ws + 134217728);
    ushort_t* Wlo = (ushort_t*)(ws + 147324928);
    float*    P   = (float*)(ws + 160432128);
    float*    G   = (float*)(ws + 162004992);

    const int nX = BB * DD;
    k_split_x<<<(nX + 255) / 256, 256, 0, stream>>>(X0, Xhi, Xlo, nX);
    k_build_w<<<NP, 256, 0, stream>>>(U, V, Wg, Whi, Wlo);

    dim3 grid(BB / 64, NGRP);
    k_gemm<<<grid, 256, 0, stream>>>(Xhi, Xlo, Whi, Wlo, P, G);

    k_final<<<BB / 8, 256, 0, stream>>>(X0, P, G, bg, out);
}
